// non_local_attention_5411658793566
// MI455X (gfx1250) — matrix operand order repaired and checked by reading, compiled, not run on hardware
//
#include <hip/hip_runtime.h>

typedef __attribute__((ext_vector_type(16))) _Float16 v16h;
typedef __attribute__((ext_vector_type(8)))  float    v8f;

#define B_SZ   8
#define NCIN   128
#define N_PIX  4096
#define C_K    64
#define C_V    64
#define C_O    128

#if __has_builtin(__builtin_amdgcn_tensor_load_to_lds) && \
    __has_builtin(__builtin_amdgcn_s_wait_tensorcnt)
#define HAVE_TDM 1
#else
#define HAVE_TDM 0
#endif

// ---------- WMMA fragment helpers (CDNA5 16x16x32 f16, wave32) ----------

// A: 16x32 tile, row-major source with leading dim lda (halfs, row = 16B-aligned).
// Lane L: m = L&15 ; lanes>=16 hold K-halves 16..31.
__device__ inline v16h frag_a(const _Float16* base, int lda) {
  int lane = threadIdx.x & 31;
  int m  = lane & 15;
  int kh = (lane & 16) ? 16 : 0;
  const _Float16* p = base + m * lda + kh;
  v16h a;
#pragma unroll
  for (int j = 0; j < 16; ++j) { a[j] = p[j]; }
  return a;
}

// B staged pre-swizzled: 32x16 tile = 512 halfs; lane L owns 16 contiguous
// halfs = column n = L&15 of K-rows ks..ks+15 (ks = 16*(L>=16)).
template <int NT>
__device__ inline void stage_b(_Float16* sB, int k, int n, _Float16 val) {
  int lane = (k & 16) | (n & 15);
  int idx  = (((k >> 5) * NT + (n >> 4)) << 9) + (lane << 4) + (k & 15);
  sB[idx] = val;
}

// Fragment read = one 32-byte contiguous per-lane load (2x ds_load_b128).
__device__ inline v16h frag_b_swz(const _Float16* tile) {
  int lane = threadIdx.x & 31;
  return *(const v16h*)(tile + (lane << 4));
}

// C/D: VGPR r holds row m = r + 8*(lane>=16), col n = lane&15.
__device__ inline void store_c(float* out, int ldo, v8f c) {
  int lane = threadIdx.x & 31;
  int n  = lane & 15;
  int mo = (lane & 16) ? 8 : 0;
#pragma unroll
  for (int r = 0; r < 8; ++r) out[(r + mo) * ldo + n] = c[r];
}

__device__ inline v8f wmma16(v16h a, v16h b, v8f c) {
  return __builtin_amdgcn_wmma_f32_16x16x32_f16(false, a, false, b, (short)0, c,
                                                false, false);
}

#if HAVE_TDM
// ---------- TDM: DMA one 2D f16 tile Global->LDS (row-major, padded rows) ----
// pad codes per D# group1: interval 0=2,1=4,..,5=64 DW ; amount 0=1,..,3=4 DW
__device__ inline void tdm_load_2d_f16(unsigned lds_byte_off,
                                       const _Float16* gptr,
                                       unsigned tile_d0, unsigned tile_d1,
                                       unsigned tensor_d0, unsigned tensor_d1,
                                       unsigned stride_d0,
                                       unsigned pad_int_code, unsigned pad_amt_code) {
  typedef __attribute__((ext_vector_type(4))) unsigned int u32x4;
  typedef __attribute__((ext_vector_type(8))) int i32x8;
  typedef __attribute__((ext_vector_type(4))) int i32x4;
  unsigned long long ga = (unsigned long long)(size_t)gptr;
  u32x4 g0;
  g0[0] = 1u;                                            // count=1 (valid user D#)
  g0[1] = lds_byte_off;                                  // lds_addr
  g0[2] = (unsigned)(ga & 0xffffffffu);                  // global_addr[31:0]
  g0[3] = (unsigned)((ga >> 32) & 0x01ffffffu) | (2u << 30); // addr[56:32] | type=2
  i32x8 g1;
  g1[0] = (int)((1u << 16) | (1u << 20) |                // data_size=2B, pad_enable
                (pad_int_code << 22) | (pad_amt_code << 25));
  g1[1] = (int)(tensor_d0 << 16);                        // tensor_dim0 @ bits 79:48
  g1[2] = (int)((tensor_d0 >> 16) | (tensor_d1 << 16));  // + tensor_dim1 @ 111:80
  g1[3] = (int)((tensor_d1 >> 16) | (tile_d0 << 16));    // + tile_dim0  @ 127:112
  g1[4] = (int)(tile_d1 & 0xffffu);                      // tile_dim1 ; tile_dim2=0
  g1[5] = (int)stride_d0;                                // tensor_dim0_stride[31:0]
  g1[6] = 0;
  g1[7] = 0;
  i32x4 z4 = {0, 0, 0, 0};
  i32x8 z8 = {0, 0, 0, 0, 0, 0, 0, 0};
  __builtin_amdgcn_tensor_load_to_lds(g0, g1, z4, z4, z8, 0);
}
#endif

// ---------- kernel 0: zero accumulators ----------
__global__ void k_zero(float* p, int n) {
  int i = blockIdx.x * blockDim.x + threadIdx.x;
  if (i < n) p[i] = 0.0f;
}

// ---------- kernel 1: 1x1 conv GEMM  y = W @ x  (+ BN stats / vsum) ----------
// grid (N/64, B, 3): z=0 query->yq, z=1 key->yk, z=2 value->v16 (f16, [N][64])
__global__ __launch_bounds__(256) void k_conv(
    const float* __restrict__ q_in, const float* __restrict__ k_in,
    const float* __restrict__ v_in,
    const float* __restrict__ Wk, const float* __restrict__ bk,
    const float* __restrict__ Wv, const float* __restrict__ bv,
    float* __restrict__ yq, float* __restrict__ yk,
    _Float16* __restrict__ v16, float* __restrict__ stats,
    float* __restrict__ vsum) {
  __shared__ _Float16 sW[64 * 136];               // A, padded lda
  __shared__ __align__(32) _Float16 sX[128 * 64]; // B, swizzled (NT=4)
  __shared__ float    sY[64 * 65];

  int tid = threadIdx.x;
  int tensor = blockIdx.z;
  int b  = blockIdx.y;
  int n0 = blockIdx.x * 64;

  const float* X    = (tensor == 0) ? q_in : (tensor == 1) ? k_in : v_in;
  const float* W    = (tensor == 2) ? Wv : Wk;
  const float* bias = (tensor == 2) ? bv : bk;

  for (int i = tid; i < 64 * 128; i += 256) {
    int m = i >> 7, kq = i & 127;
    sW[m * 136 + kq] = (_Float16)W[i];
  }
  const float* Xb = X + (size_t)b * NCIN * N_PIX + n0;
  for (int i = tid; i < 128 * 64; i += 256) {
    int c = i >> 6, j = i & 63;
    stage_b<4>(sX, c, j, (_Float16)Xb[(size_t)c * N_PIX + j]);
  }
  __syncthreads();

  int w   = tid >> 5;
  int m0  = (w & 3) * 16;
  int nt0 = (w >> 2) * 2;  // 2 n-tiles per wave
  v8f acc0 = {}; v8f acc1 = {};
#pragma unroll
  for (int kk = 0; kk < 4; ++kk) {   // K = 128 = 4 * 32
    v16h a  = frag_a(&sW[m0 * 136 + kk * 32], 136);
    v16h b0 = frag_b_swz(&sX[((kk << 2) + nt0) << 9]);
    v16h b1 = frag_b_swz(&sX[((kk << 2) + nt0 + 1) << 9]);
    acc0 = wmma16(a, b0, acc0);
    acc1 = wmma16(a, b1, acc1);
  }
  store_c(&sY[m0 * 65 + nt0 * 16], 65, acc0);
  store_c(&sY[m0 * 65 + nt0 * 16 + 16], 65, acc1);
  __syncthreads();

  if (tensor < 2) {
    float* Y = ((tensor == 0) ? yq : yk) + (size_t)b * C_K * N_PIX + n0;
    for (int i = tid; i < 64 * 64; i += 256) {
      int c = i >> 6, j = i & 63;
      Y[(size_t)c * N_PIX + j] = sY[c * 65 + j] + bias[c];
    }
    if (tid < 64) {
      int c = tid; float bb = bias[c];
      float s = 0.0f, ss = 0.0f;
      for (int j = 0; j < 64; ++j) {
        float yv = sY[c * 65 + j] + bb; s += yv; ss += yv * yv;
      }
      unsafeAtomicAdd(&stats[tensor * 128 + c], s);
      unsafeAtomicAdd(&stats[tensor * 128 + 64 + c], ss);
    }
  } else {
    _Float16* Vb = v16 + (size_t)b * N_PIX * C_V + (size_t)n0 * C_V;
    int c = tid & 63;
    float bb = bias[c];
    for (int j = tid >> 6; j < 64; j += 4)
      Vb[(size_t)j * 64 + c] = (_Float16)(sY[c * 65 + j] + bb);
    if (tid < 64) {
      float s = 0.0f; float bc = bias[tid];
      for (int j = 0; j < 64; ++j) s += sY[tid * 65 + j] + bc;
      unsafeAtomicAdd(&vsum[b * 64 + tid], s);
    }
  }
}

// ---------- kernel 2: finalize BN scale/shift ----------
__global__ void k_bnstats(const float* __restrict__ stats,
                          const float* __restrict__ gamma,
                          const float* __restrict__ beta,
                          float* __restrict__ bnp) {
  int t = threadIdx.x;           // 0..127
  int tensor = t >> 6, c = t & 63;
  float cnt  = (float)(B_SZ * N_PIX);
  float mean = stats[tensor * 128 + c] / cnt;
  float var  = stats[tensor * 128 + 64 + c] / cnt - mean * mean;
  float sc   = gamma[c] * rsqrtf(var + 1e-5f);
  bnp[tensor * 128 + c]      = sc;
  bnp[tensor * 128 + 64 + c] = beta[c] - mean * sc;
}

// ---------- kernel 3: BN + per-pixel L2 normalize -> f16 [64][N] ----------
__global__ __launch_bounds__(256) void k_norm(
    const float* __restrict__ yq, const float* __restrict__ yk,
    const float* __restrict__ bnp,
    _Float16* __restrict__ qn, _Float16* __restrict__ kn) {
  int tensor = blockIdx.z;
  int b = blockIdx.y;
  int n = blockIdx.x * 256 + threadIdx.x;
  const float* Y = ((tensor == 0) ? yq : yk) + (size_t)b * C_K * N_PIX + n;
  _Float16*   O  = ((tensor == 0) ? qn : kn) + (size_t)b * C_K * N_PIX + n;
  const float* sc = bnp + tensor * 128;
  float ss = 0.0f;
  for (int c = 0; c < 64; ++c) {
    float t = Y[(size_t)c * N_PIX] * sc[c] + sc[64 + c];
    ss += t * t;
  }
  float inv = 1.0f / (sqrtf(ss) + 1e-7f);
  for (int c = 0; c < 64; ++c) {
    float t = Y[(size_t)c * N_PIX] * sc[c] + sc[64 + c];
    O[(size_t)c * N_PIX] = (_Float16)(t * inv);
  }
}

// ---------- kernel 4: KV[b] = kn[b] (64xN) @ v[b] (Nx64), split-K ----------
__global__ __launch_bounds__(256) void k_kv(
    const _Float16* __restrict__ kn, const _Float16* __restrict__ v16,
    float* __restrict__ KV) {
  __shared__ _Float16 sK[64 * 136];               // A, padded lda (TDM pad)
  __shared__ __align__(32) _Float16 sV[128 * 64]; // B, swizzled (NT=4)
  int tid = threadIdx.x;
  int b   = blockIdx.y;
  int k0  = blockIdx.x * 256;   // this block's K chunk
  const _Float16* Kb = kn  + (size_t)b * C_K * N_PIX;
  const _Float16* Vb = v16 + (size_t)b * N_PIX * C_V;
  int w   = tid >> 5;
  int m0  = (w & 3) * 16;
  int nt0 = (w >> 2) * 2;
  v8f acc0 = {}; v8f acc1 = {};
#if HAVE_TDM
  unsigned sK_off = (unsigned)(size_t)(void*)sK;  // LDS byte offset (flat low 32b)
#endif
  for (int sub = 0; sub < 2; ++sub) {
    int kb = k0 + sub * 128;
    __syncthreads();
#if HAVE_TDM
    // DMA the 64x128 f16 K-tile into LDS with 8-half row padding (stride 136).
    if (w == 0)
      tdm_load_2d_f16(sK_off, Kb + kb, /*tile_d0=*/128, /*tile_d1=*/64,
                      /*tensor_d0=*/N_PIX, /*tensor_d1=*/64,
                      /*stride_d0=*/N_PIX, /*pad_int=64DW*/5, /*pad_amt=4DW*/3);
#else
    for (int i = tid; i < 64 * 128; i += 256) {
      int c = i >> 7, j = i & 127;
      sK[c * 136 + j] = Kb[(size_t)c * N_PIX + kb + j];
    }
#endif
    {
      int c = tid & 63;
      for (int j = tid >> 6; j < 128; j += 4)
        stage_b<4>(sV, j, c, Vb[(size_t)(kb + j) * 64 + c]);
    }
#if HAVE_TDM
    if (w == 0) __builtin_amdgcn_s_wait_tensorcnt(0);
#endif
    __syncthreads();
#pragma unroll
    for (int kk = 0; kk < 4; ++kk) {
      v16h a  = frag_a(&sK[m0 * 136 + kk * 32], 136);
      v16h b0 = frag_b_swz(&sV[((kk << 2) + nt0) << 9]);
      v16h b1 = frag_b_swz(&sV[((kk << 2) + nt0 + 1) << 9]);
      acc0 = wmma16(a, b0, acc0);
      acc1 = wmma16(a, b1, acc1);
    }
  }
  int lane = tid & 31;
  int n  = lane & 15;
  int mo = (lane & 16) ? 8 : 0;
  float* KVb = KV + (size_t)b * 64 * 64;
#pragma unroll
  for (int r = 0; r < 8; ++r) {
    unsafeAtomicAdd(&KVb[(m0 + r + mo) * 64 + nt0 * 16 + n],      acc0[r]);
    unsafeAtomicAdd(&KVb[(m0 + r + mo) * 64 + nt0 * 16 + 16 + n], acc1[r]);
  }
}

// ---------- kernel 5: E[b] = Ww @ KV[b]^T ; f[b] = Ww @ vsum[b] + bw ----------
__global__ __launch_bounds__(256) void k_ef(
    const float* __restrict__ KV, const float* __restrict__ vsum,
    const float* __restrict__ Ww, const float* __restrict__ bw,
    _Float16* __restrict__ E, float* __restrict__ fvec) {
  __shared__ float sKV[64 * 64];
  __shared__ float sVS[64];
  int tid = threadIdx.x;
  int b   = blockIdx.x;
  for (int i = tid; i < 4096; i += 256) sKV[i] = KV[(size_t)b * 4096 + i];
  if (tid < 64) sVS[tid] = vsum[b * 64 + tid];
  __syncthreads();
  for (int i = tid; i < 128 * 64; i += 256) {
    int o = i >> 6, c = i & 63;
    float s = 0.0f;
    for (int vv = 0; vv < 64; ++vv) s += Ww[o * 64 + vv] * sKV[c * 64 + vv];
    E[(size_t)b * 8192 + o * 64 + c] = (_Float16)s;
  }
  if (tid < 128) {
    float s = bw[tid];
    for (int vv = 0; vv < 64; ++vv) s += Ww[tid * 64 + vv] * sVS[vv];
    fvec[b * 128 + tid] = s;
  }
}

// ---------- kernel 6: out[b] = E[b] (128x64) @ qn[b] (64xN) + f[b] ----------
__global__ __launch_bounds__(256) void k_out(
    const _Float16* __restrict__ E, const _Float16* __restrict__ qn,
    const float* __restrict__ fvec, float* __restrict__ out) {
  __shared__ _Float16 sE[128 * 72];              // A, padded lda (TDM pad)
  __shared__ __align__(32) _Float16 sQ[64 * 64]; // B, swizzled (NT=4)
  int tid = threadIdx.x;
  int b   = blockIdx.y;
  int n0  = blockIdx.x * 64;
  int w   = tid >> 5;
#if HAVE_TDM
  // DMA E[b] (128x64 f16) into LDS with 8-half row padding (stride 72).
  if (w == 0)
    tdm_load_2d_f16((unsigned)(size_t)(void*)sE, E + (size_t)b * 8192,
                    /*tile_d0=*/64, /*tile_d1=*/128,
                    /*tensor_d0=*/64, /*tensor_d1=*/128,
                    /*stride_d0=*/64, /*pad_int=32DW*/4, /*pad_amt=4DW*/3);
#else
  for (int i = tid; i < 128 * 64; i += 256) {
    int o = i >> 6, c = i & 63;
    sE[o * 72 + c] = E[(size_t)b * 8192 + o * 64 + c];
  }
#endif
  const _Float16* Qb = qn + (size_t)b * C_K * N_PIX + n0;
  for (int i = tid; i < 64 * 64; i += 256) {
    int c = i >> 6, j = i & 63;
    stage_b<4>(sQ, c, j, Qb[(size_t)c * N_PIX + j]);
  }
#if HAVE_TDM
  if (w == 0) __builtin_amdgcn_s_wait_tensorcnt(0);
#endif
  __syncthreads();
  int m0 = w * 16;                 // 8 waves cover M=128
  v8f acc[4] = {v8f{}, v8f{}, v8f{}, v8f{}};
#pragma unroll
  for (int kk = 0; kk < 2; ++kk) { // K = 64
    v16h a = frag_a(&sE[m0 * 72 + kk * 32], 72);
#pragma unroll
    for (int nt = 0; nt < 4; ++nt) {
      v16h bf = frag_b_swz(&sQ[((kk << 2) + nt) << 9]);
      acc[nt] = wmma16(a, bf, acc[nt]);
    }
  }
  int lane = tid & 31;
  int n  = lane & 15;
  int mo = (lane & 16) ? 8 : 0;
  float* Ob = out + (size_t)b * C_O * N_PIX + n0;
#pragma unroll
  for (int r = 0; r < 8; ++r) {
    int m = m0 + r + mo;
    float fv = fvec[b * 128 + m];
#pragma unroll
    for (int nt = 0; nt < 4; ++nt)
      Ob[(size_t)m * N_PIX + nt * 16 + n] = acc[nt][r] + fv;
  }
}

extern "C" void kernel_launch(void* const* d_in, const int* in_sizes, int n_in,
                              void* d_out, int out_size, void* d_ws, size_t ws_size,
                              hipStream_t stream) {
  const float* q_in  = (const float*)d_in[0];
  const float* k_in  = (const float*)d_in[1];
  const float* v_in  = (const float*)d_in[2];
  const float* Wk    = (const float*)d_in[3];
  const float* bk    = (const float*)d_in[4];
  const float* gamma = (const float*)d_in[5];
  const float* beta  = (const float*)d_in[6];
  const float* Wv    = (const float*)d_in[7];
  const float* bv    = (const float*)d_in[8];
  const float* Ww    = (const float*)d_in[9];
  const float* bw    = (const float*)d_in[10];
  float* out = (float*)d_out;

  char* ws = (char*)d_ws;
  float*    yq    = (float*)(ws + 0);            // 8 MiB  [B][64][N] f32
  float*    yk    = (float*)(ws + 8388608);      // 8 MiB
  _Float16* v16   = (_Float16*)(ws + 16777216);  // 4 MiB  [B][N][64] f16
  _Float16* qn    = (_Float16*)(ws + 20971520);  // 4 MiB  [B][64][N] f16
  _Float16* kn    = (_Float16*)(ws + 25165824);  // 4 MiB
  float*    stats = (float*)(ws + 29360128);     // [2][2][64]
  float*    bnp   = (float*)(ws + 29361152);     // [2][2][64]
  float*    vsum  = (float*)(ws + 29362176);     // [B][64]
  float*    KV    = (float*)(ws + 29364224);     // [B][64][64] f32
  _Float16* E     = (_Float16*)(ws + 29495296);  // [B][128][64] f16
  float*    fvec  = (float*)(ws + 29626368);     // [B][128]

  // zero stats..KV (contiguous region: 135168 bytes = 33792 floats)
  int zn = 33792;
  k_zero<<<(zn + 255) / 256, 256, 0, stream>>>(stats, zn);

  k_conv<<<dim3(N_PIX / 64, B_SZ, 3), 256, 0, stream>>>(
      q_in, k_in, v_in, Wk, bk, Wv, bv, yq, yk, v16, stats, vsum);
  k_bnstats<<<1, 128, 0, stream>>>(stats, gamma, beta, bnp);
  k_norm<<<dim3(N_PIX / 256, B_SZ, 2), 256, 0, stream>>>(yq, yk, bnp, qn, kn);
  k_kv<<<dim3(16, B_SZ), 256, 0, stream>>>(kn, v16, KV);
  k_ef<<<B_SZ, 256, 0, stream>>>(KV, vsum, Ww, bw, E, fvec);
  k_out<<<dim3(N_PIX / 64, B_SZ), 256, 0, stream>>>(E, qn, fvec, out);
}
